// MultiHeadAttentionXL_63015760167568
// MI455X (gfx1250) — matrix operand order repaired and checked by reading, not compiled, not run on hardware
//
#include <hip/hip_runtime.h>

// ---------------------------------------------------------------------------
// CDNA5 (gfx1250) Transformer-XL attention, bf16 WMMA everywhere.
// GEMMs stage B tiles in LDS via global_load_async_to_lds_b128 (double-buffer).
// ---------------------------------------------------------------------------

typedef __bf16 bf16;
typedef __attribute__((ext_vector_type(8)))  __bf16 bf16x8;
typedef __attribute__((ext_vector_type(16))) __bf16 bf16x16;
typedef __attribute__((ext_vector_type(8)))  float  f32x8;

#define T_SEQ   2048
#define D_MODEL 1024
#define N_HEADS 16
#define D_HEAD  64

__device__ __forceinline__ f32x8 wmma_bf16(bf16x16 a, bf16x16 b, f32x8 c) {
  // D = A(16x32) x B(32x16) + C, fp32 accumulate
  return __builtin_amdgcn_wmma_f32_16x16x32_bf16(false, a, false, b, (short)0, c,
                                                 false, false);
}

// A-matrix fragment (16x32, 16-bit): lane m = lane%16 holds row m.
// elements 0..15 = K (16*half + 0..15) (contiguous per lane).
__device__ __forceinline__ bf16x16 load_frag_A(const bf16* __restrict__ base,
                                               int stride, int row, int kbase) {
  const int lane = threadIdx.x & 31;
  const int half = lane >> 4;
  const bf16* p = base + (size_t)row * stride + kbase + 16 * half;
  bf16x8 lo = *(const bf16x8*)(p);
  bf16x8 hi = *(const bf16x8*)(p + 8);
  return __builtin_shufflevector(lo, hi, 0,1,2,3,4,5,6,7,8,9,10,11,12,13,14,15);
}

// B-matrix fragment (32x16, 16-bit): lane n = lane%16 holds column n.
// lanes 0-15 hold K=0..15, lanes 16-31 hold K=16..31 (contiguous per lane).
// "row" is the B-column index into a [N][K]-stored (i.e. transposed) matrix.
__device__ __forceinline__ bf16x16 load_frag_B(const bf16* __restrict__ base,
                                               int stride, int row, int kbase) {
  const int lane = threadIdx.x & 31;
  const int half = lane >> 4;
  const bf16* p = base + (size_t)row * stride + kbase + 16 * half;
  bf16x8 lo = *(const bf16x8*)(p);
  bf16x8 hi = *(const bf16x8*)(p + 8);
  return __builtin_shufflevector(lo, hi, 0,1,2,3,4,5,6,7,8,9,10,11,12,13,14,15);
}

// Same B-fragment pattern but reading a [rows][32]-bf16 LDS tile.
__device__ __forceinline__ bf16x16 load_frag_B_lds(const bf16* p) {
  bf16x8 lo = *(const bf16x8*)(p);
  bf16x8 hi = *(const bf16x8*)(p + 8);
  return __builtin_shufflevector(lo, hi, 0,1,2,3,4,5,6,7,8,9,10,11,12,13,14,15);
}

// ---------------------------------------------------------------------------
// fp32 -> bf16 converts
// ---------------------------------------------------------------------------
__global__ void cvt_bf16_kernel(const float* __restrict__ s, bf16* __restrict__ d, int n) {
  for (int i = blockIdx.x * blockDim.x + threadIdx.x; i < n; i += gridDim.x * blockDim.x)
    d[i] = (bf16)s[i];
}

// dst[c*R + r] = (bf16) src[r*C + c]   (transpose + convert; coalesced reads)
__global__ void cvt_bf16_T_kernel(const float* __restrict__ s, bf16* __restrict__ d,
                                  int R, int C) {
  const int n = R * C;
  for (int i = blockIdx.x * blockDim.x + threadIdx.x; i < n; i += gridDim.x * blockDim.x) {
    int r = i / C, c = i - r * C;
    d[(size_t)c * R + r] = (bf16)s[i];
  }
}

// ---------------------------------------------------------------------------
// Generic WMMA GEMM: C[M,N] = A[M,K] (row-major bf16) @ B, with B supplied
// TRANSPOSED as Bt[N,K] row-major bf16. The 64x32 B k-slice is staged in LDS
// via async DMA (global_load_async_to_lds_b128), double-buffered: 4 KB/slice,
// 2 async b128 per thread per slice (ASYNCcnt += 2 per wave per slice).
// MODE 0: bf16 row-major out (+ optional fp32 bias[N])
// MODE 1: bf16 transposed out, Ct[N][M] (packed 16B stores along M)
// MODE 2: fp32 row-major out
// Block = 128 threads = 4 waves; each wave does a 16(M) x 64(N) tile.
// ---------------------------------------------------------------------------
template <int MODE, bool BIAS>
__global__ __launch_bounds__(128) void gemm_wmma_kernel(
    const bf16* __restrict__ A, const bf16* __restrict__ Bt, void* __restrict__ Cv,
    const float* __restrict__ bias, int M, int N, int K) {
  __shared__ __align__(16) bf16 Bs[2][64][32];   // double-buffered 64x32 slice

  const int tid  = threadIdx.x;
  const int lane = tid & 31;
  const int wv   = tid >> 5;
  const int half = lane >> 4;
  const int c16  = lane & 15;
  const int m0   = blockIdx.x * 64 + wv * 16;
  const int n0   = blockIdx.y * 64;

  // async-staging assignment: 256 x 16B chunks, 2 per thread
  const int c0 = tid, c1 = tid + 128;
  const int row0 = c0 >> 2, off0 = (c0 & 3) * 8;   // row in [0,64), off in halves
  const int row1 = c1 >> 2, off1 = (c1 & 3) * 8;
  const bf16* gb0 = Bt + (size_t)(n0 + row0) * K + off0;
  const bf16* gb1 = Bt + (size_t)(n0 + row1) * K + off1;

  auto issue_async = [&](int k0, int buf) {
    unsigned l0 = (unsigned)(size_t)&Bs[buf][row0][off0];  // LDS byte offset
    unsigned l1 = (unsigned)(size_t)&Bs[buf][row1][off1];
    unsigned long long g0 = (unsigned long long)(size_t)(gb0 + k0);
    unsigned long long g1 = (unsigned long long)(size_t)(gb1 + k0);
    asm volatile("global_load_async_to_lds_b128 %0, %1, off"
                 :: "v"(l0), "v"(g0) : "memory");
    asm volatile("global_load_async_to_lds_b128 %0, %1, off"
                 :: "v"(l1), "v"(g1) : "memory");
  };

  const f32x8 zero = {0.f, 0.f, 0.f, 0.f, 0.f, 0.f, 0.f, 0.f};
  f32x8 acc[4];
#pragma unroll
  for (int i = 0; i < 4; ++i) acc[i] = zero;

  issue_async(0, 0);
  int buf = 0;
  for (int k0 = 0; k0 < K; k0 += 32, buf ^= 1) {
    if (k0 + 32 < K) {
      issue_async(k0 + 32, buf ^ 1);
      asm volatile("s_wait_asynccnt 2" ::: "memory");  // current slice landed
    } else {
      asm volatile("s_wait_asynccnt 0" ::: "memory");
    }
    __syncthreads();  // all waves' slice chunks visible

    if (k0 + 256 < K)  // gentle L2 prefetch of upcoming A row data
      __builtin_prefetch(A + (size_t)(m0 + c16) * K + k0 + 256, 0, 0);
    bf16x16 af = load_frag_A(A, K, m0 + c16, k0);
#pragma unroll
    for (int nt = 0; nt < 4; ++nt) {
      bf16x16 bfg = load_frag_B_lds(&Bs[buf][nt * 16 + c16][16 * half]);
      acc[nt] = wmma_bf16(af, bfg, acc[nt]);
    }
    __syncthreads();  // reads done before buffer is refilled
  }

#pragma unroll
  for (int nt = 0; nt < 4; ++nt) {
    const int col = n0 + nt * 16 + c16;
    if constexpr (MODE == 1) {
      bf16x8 pk;
#pragma unroll
      for (int p = 0; p < 8; ++p) pk[p] = (bf16)acc[nt][p];
      bf16* Ct = (bf16*)Cv;  // rows m0+8*half .. +7 are contiguous along M
      *(bf16x8*)(Ct + (size_t)col * M + m0 + 8 * half) = pk;
    } else {
      float bv = 0.f;
      if constexpr (BIAS) bv = bias[col];
#pragma unroll
      for (int p = 0; p < 8; ++p) {
        const int row = m0 + p + 8 * half;
        const float v = acc[nt][p] + bv;
        if constexpr (MODE == 0) ((bf16*)Cv)[(size_t)row * N + col] = (bf16)v;
        else                     ((float*)Cv)[(size_t)row * N + col] = v;
      }
    }
  }
}

// ---------------------------------------------------------------------------
// Flash-style causal XL attention. Grid: (T/64, H). Block = 4 waves.
// Wave w owns query rows qi..qi+15 (qi = blockIdx.x*64 + w*16); iterates keys
// in groups of 32. Per group: AC via 4 WMMA, BD via 8 WMMA (two shifted
// r-windows, diagonal gather through LDS), online softmax, P@V via 4 WMMA.
// No block-wide barriers (waves have different trip counts); per-wave LDS
// regions + in-order DS + __threadfence_block() for ordering.
// ---------------------------------------------------------------------------
__global__ __launch_bounds__(128) void flash_xl_kernel(
    const bf16* __restrict__ qu, const bf16* __restrict__ qv,
    const bf16* __restrict__ kk, const bf16* __restrict__ vT,
    const bf16* __restrict__ rp, bf16* __restrict__ outbf) {
  constexpr int T = T_SEQ, D = D_MODEL, DH = D_HEAD;
  __shared__ float lds_g[4][16][32];                 // G0|G1 windows per wave
  __shared__ __align__(16) bf16 lds_p[4][16][32];    // P tile (A-layout staging)

  const int lane = threadIdx.x & 31;
  const int wv   = threadIdx.x >> 5;
  const int half = lane >> 4;
  const int c16  = lane & 15;
  const int h    = blockIdx.y;
  const int hd0  = h * DH;
  const int qi   = blockIdx.x * 64 + wv * 16;

  // Q fragments (with u / v_bias already fused), loaded once.
  bf16x16 quf[2], qvf[2];
#pragma unroll
  for (int ks = 0; ks < 2; ++ks) {
    quf[ks] = load_frag_A(qu, D, qi + c16, hd0 + 32 * ks);
    qvf[ks] = load_frag_A(qv, D, qi + c16, hd0 + 32 * ks);
  }

  const f32x8 zero = {0.f, 0.f, 0.f, 0.f, 0.f, 0.f, 0.f, 0.f};
  f32x8 O[4];
#pragma unroll
  for (int i = 0; i < 4; ++i) O[i] = zero;
  float mrow[8], lrow[8];
#pragma unroll
  for (int p = 0; p < 8; ++p) { mrow[p] = -1e30f; lrow[p] = 0.f; }

  for (int jj = 0; jj <= qi + 15; jj += 32) {  // jj <= 2016 -> all loads in range
    f32x8 S[2];
#pragma unroll
    for (int jt2 = 0; jt2 < 2; ++jt2) {
      const int jtile = jj + jt2 * 16;
      // --- AC = (q+u) . k ---
      f32x8 sac = zero;
#pragma unroll
      for (int ks = 0; ks < 2; ++ks) {
        bf16x16 kf = load_frag_B(kk, D, jtile + c16, hd0 + 32 * ks);
        sac = wmma_bf16(quf[ks], kf, sac);
      }
      // --- BD windows: BD(a,b) = (q+v)_a . r_{cbase - a + b} ---
      const int cbase = T - 1 - qi + jtile;
      int r0 = cbase - 15 + c16; r0 = r0 < 0 ? 0 : (r0 > T - 1 ? T - 1 : r0);
      int r1 = cbase + 1 + c16;  r1 = r1 > T - 1 ? T - 1 : r1;
      f32x8 g0 = zero, g1 = zero;
#pragma unroll
      for (int ks = 0; ks < 2; ++ks) {
        bf16x16 rf0 = load_frag_B(rp, D, r0, hd0 + 32 * ks);
        g0 = wmma_bf16(qvf[ks], rf0, g0);
        bf16x16 rf1 = load_frag_B(rp, D, r1, hd0 + 32 * ks);
        g1 = wmma_bf16(qvf[ks], rf1, g1);
      }
      // diagonal gather Gfull[a, 15-a+b] through per-wave LDS
#pragma unroll
      for (int p = 0; p < 8; ++p) {
        const int a = p + 8 * half;
        lds_g[wv][a][c16]      = g0[p];
        lds_g[wv][a][16 + c16] = g1[p];
      }
      __threadfence_block();
#pragma unroll
      for (int p = 0; p < 8; ++p) {
        const int a = p + 8 * half;
        const float bd = lds_g[wv][a][15 - a + c16];
        const float s  = (sac[p] + bd) * 0.125f;          // 1/sqrt(64)
        const bool ok  = (jtile + c16) <= (qi + a);       // causal mask
        S[jt2][p] = ok ? s : -1e30f;
      }
      __threadfence_block();
    }

    // --- online softmax over these 32 key columns ---
#pragma unroll
    for (int p = 0; p < 8; ++p) {
      float tmax = fmaxf(S[0][p], S[1][p]);
#pragma unroll
      for (int off = 1; off < 16; off <<= 1)
        tmax = fmaxf(tmax, __shfl_xor(tmax, off, 32));
      const float nm = fmaxf(mrow[p], tmax);
      const float sc = __expf(mrow[p] - nm);
      const float e0 = __expf(S[0][p] - nm);
      const float e1 = __expf(S[1][p] - nm);
      S[0][p] = e0; S[1][p] = e1;
      float rs = e0 + e1;
#pragma unroll
      for (int off = 1; off < 16; off <<= 1) rs += __shfl_xor(rs, off, 32);
      lrow[p] = lrow[p] * sc + rs;
      mrow[p] = nm;
#pragma unroll
      for (int nt = 0; nt < 4; ++nt) O[nt][p] *= sc;
    }

    // --- P (C-layout) -> LDS row-major [16][32] -> A-layout fragment ---
#pragma unroll
    for (int jt2 = 0; jt2 < 2; ++jt2)
#pragma unroll
      for (int p = 0; p < 8; ++p) {
        const int a = p + 8 * half;
        lds_p[wv][a][jt2 * 16 + c16] = (bf16)S[jt2][p];
      }
    __threadfence_block();
    bf16x8 plo = *(const bf16x8*)&lds_p[wv][c16][16 * half];
    bf16x8 phi = *(const bf16x8*)&lds_p[wv][c16][16 * half + 8];
    bf16x16 pfrag = __builtin_shufflevector(plo, phi,
        0,1,2,3,4,5,6,7,8,9,10,11,12,13,14,15);
    __threadfence_block();

    // --- O += P @ V   (V stored transposed: vT[d, t]) ---
#pragma unroll
    for (int nt = 0; nt < 4; ++nt) {
      bf16x16 vf = load_frag_B(vT, T, hd0 + nt * 16 + c16, jj);
      O[nt] = wmma_bf16(pfrag, vf, O[nt]);
    }
  }

  // epilogue: normalize, store bf16 for the Wo GEMM
#pragma unroll
  for (int nt = 0; nt < 4; ++nt) {
    const int col = hd0 + nt * 16 + c16;
#pragma unroll
    for (int p = 0; p < 8; ++p) {
      const int row = qi + p + 8 * half;
      outbf[(size_t)row * D + col] = (bf16)(O[nt][p] * (1.0f / lrow[p]));
    }
  }
}

// ---------------------------------------------------------------------------
// Host-side launch
// ---------------------------------------------------------------------------
extern "C" void kernel_launch(void* const* d_in, const int* in_sizes, int n_in,
                              void* d_out, int out_size, void* d_ws, size_t ws_size,
                              hipStream_t stream) {
  (void)in_sizes; (void)n_in; (void)out_size; (void)ws_size;
  constexpr int T = T_SEQ, D = D_MODEL;

  const float* x     = (const float*)d_in[0];
  const float* r_emb = (const float*)d_in[1];
  const float* Wq    = (const float*)d_in[2];
  const float* Wk    = (const float*)d_in[3];
  const float* Wv    = (const float*)d_in[4];
  const float* Wr    = (const float*)d_in[5];
  const float* Wo    = (const float*)d_in[6];
  const float* u     = (const float*)d_in[7];  // [H,DH] flat == length-D bias
  const float* vb    = (const float*)d_in[8];

  char* ws = (char*)d_ws;
  size_t off = 0;
  auto take = [&](size_t elems) -> bf16* {
    bf16* p = (bf16*)(ws + off);
    off += (elems * sizeof(bf16) + 255) & ~(size_t)255;
    return p;
  };
  bf16* x_bf = take((size_t)T * D);
  bf16* r_bf = take((size_t)T * D);
  bf16* WqT  = take((size_t)D * D);
  bf16* WkT  = take((size_t)D * D);
  bf16* WvT  = take((size_t)D * D);
  bf16* WrT  = take((size_t)D * D);
  bf16* WoT  = take((size_t)D * D);
  bf16* qu   = take((size_t)T * D);
  bf16* qv   = take((size_t)T * D);
  bf16* kk   = take((size_t)T * D);
  bf16* vTb  = take((size_t)T * D);  // [D][T]
  bf16* rp   = take((size_t)T * D);
  bf16* attn = take((size_t)T * D);

  cvt_bf16_kernel<<<1024, 256, 0, stream>>>(x, x_bf, T * D);
  cvt_bf16_kernel<<<1024, 256, 0, stream>>>(r_emb, r_bf, T * D);
  cvt_bf16_T_kernel<<<1024, 256, 0, stream>>>(Wq, WqT, D, D);
  cvt_bf16_T_kernel<<<1024, 256, 0, stream>>>(Wk, WkT, D, D);
  cvt_bf16_T_kernel<<<1024, 256, 0, stream>>>(Wv, WvT, D, D);
  cvt_bf16_T_kernel<<<1024, 256, 0, stream>>>(Wr, WrT, D, D);
  cvt_bf16_T_kernel<<<1024, 256, 0, stream>>>(Wo, WoT, D, D);

  dim3 gg(T / 64, D / 64);
  gemm_wmma_kernel<0, true ><<<gg, 128, 0, stream>>>(x_bf, WqT, qu,   u,       T, D, D);
  gemm_wmma_kernel<0, true ><<<gg, 128, 0, stream>>>(x_bf, WqT, qv,   vb,      T, D, D);
  gemm_wmma_kernel<0, false><<<gg, 128, 0, stream>>>(x_bf, WkT, kk,   nullptr, T, D, D);
  gemm_wmma_kernel<1, false><<<gg, 128, 0, stream>>>(x_bf, WvT, vTb,  nullptr, T, D, D);
  gemm_wmma_kernel<0, false><<<gg, 128, 0, stream>>>(r_bf, WrT, rp,   nullptr, T, D, D);

  dim3 gf(T / 64, N_HEADS);
  flash_xl_kernel<<<gf, 128, 0, stream>>>(qu, qv, kk, vTb, rp, attn);

  gemm_wmma_kernel<2, false><<<gg, 128, 0, stream>>>(attn, WoT, d_out, nullptr, T, D, D);
}
